// StructuredAttentionModule_38027640438895
// MI455X (gfx1250) — matrix operand order repaired and checked by reading, not compiled, not run on hardware
//
#include <hip/hip_runtime.h>
#include <hip/hip_bf16.h>
#include <math.h>

// ---------------------------------------------------------------------------
// StructuredAttention for MI455X (gfx1250)
//   - all GEMMs: NT bf16 WMMA (v_wmma_f32_16x16x32_bf16), f32 accumulate
//   - tile staging: double-buffered Tensor Data Mover (tensor_load_to_lds)
//     with LDS row padding; DMA of tile k+1 overlaps WMMA on tile k
//   - matrix-tree inversion: fp32 Gauss-Jordan, one workgroup per batch matrix
// B=32, T=512, H=512, SEM=STRUCT=128, S2=M2=256
// ---------------------------------------------------------------------------

typedef __attribute__((ext_vector_type(16))) __bf16 v16bf;
typedef __attribute__((ext_vector_type(8)))  __bf16 v8bf;
typedef __attribute__((ext_vector_type(8)))  float  v8f;

#define SA_B   32
#define SA_T   512
#define SA_BT  (SA_B * SA_T)          // 16384
#define SA_TT  (SA_T * SA_T)          // 262144
#define SA_S   256

// GEMM tiling: 256 threads = 8 waves (4x2), tile 128(M) x 64(N) x 32(K)
#define TM 128
#define TN 64
#define TK 32
#define LDT 40                        // padded LDS row in bf16 (80B = 16 DW + 4 DW pad)

#if defined(__has_builtin)
#if __has_builtin(__builtin_amdgcn_tensor_load_to_lds)
#define SA_TDM 1
#endif
#endif
#ifndef SA_TDM
#define SA_TDM 0
#endif

#if SA_TDM
typedef unsigned int sa_u32x4 __attribute__((ext_vector_type(4)));
typedef int          sa_i32x4 __attribute__((ext_vector_type(4)));
typedef int          sa_i32x8 __attribute__((ext_vector_type(8)));

// 2D TDM tile load: rows x 32 bf16, global row stride = lda elements,
// LDS rows padded 64B->80B via pad_interval=16DW, pad_amount=4DW.
__device__ __forceinline__ void sa_tdm_load(unsigned int lds_addr,
                                            const __bf16* gptr,
                                            unsigned int rows,
                                            unsigned int lda_elems)
{
  unsigned long long ga = (unsigned long long)(uintptr_t)gptr;
  sa_u32x4 g0;
  g0[0] = 1u;                                     // count=1 (valid user D#)
  g0[1] = lds_addr;                               // bits 63:32 lds_addr
  g0[2] = (unsigned int)ga;                       // bits 95:64 global_addr lo
  g0[3] = (unsigned int)(ga >> 32) | 0x80000000u; // addr hi + type=2 (127:126)
  sa_i32x8 g1;
  g1[0] = (1 << 16)     // data_size = 1 -> 2 bytes
        | (1 << 20)     // pad_enable
        | (3 << 22)     // pad_interval = 16 DWORDs (one 32-elem bf16 row)
        | (3 << 25);    // pad_amount   = 4 DWORDs (16B) -> LDS row stride 80B
  g1[1] = (int)0xFFFF0000u;                       // tensor_dim0 lo16 (0x7FFFFFFF)
  g1[2] = (int)(0x7FFFu | 0xFFFF0000u);           // tdim0 hi16 | tdim1 lo16
  g1[3] = (int)(0x7FFFu | (32u << 16));           // tdim1 hi16 | tile_dim0 = 32
  g1[4] = (int)rows;                              // tile_dim1 = rows, tile_dim2 = 0
  g1[5] = (int)lda_elems;                         // tensor_dim0_stride lo32
  g1[6] = 0;
  g1[7] = 0;
  sa_i32x4 z4 = {0, 0, 0, 0};
#if __clang_major__ >= 23
  sa_i32x8 z8 = {0, 0, 0, 0, 0, 0, 0, 0};
  __builtin_amdgcn_tensor_load_to_lds(g0, g1, z4, z4, z8, 0);
#else
  __builtin_amdgcn_tensor_load_to_lds(g0, g1, z4, z4, 0);
#endif
}
#endif  // SA_TDM

// ---------------------------------------------------------------------------
// Generic NT bf16 GEMM:  C[M,N] = act( A(M,K) . B(N,K)^T + bias + r1row(x)r1col )
// Outputs: fp32 (Cf) and/or bf16 (Cb) and/or transposed bf16 (CbT, per-512-row
// batch layout [tb][n][t]).  M%128==0, N%64==0, K%32==0.
// ---------------------------------------------------------------------------
__global__ __launch_bounds__(256)
void sa_gemm_nt(const __bf16* __restrict__ A, long lda, long sA,
                const __bf16* __restrict__ Bw, long ldb, long sB,
                const float* __restrict__ bias,
                const float* __restrict__ r1row, const float* __restrict__ r1col,
                float* __restrict__ Cf, __bf16* __restrict__ Cb,
                __bf16* __restrict__ CbT,
                long ldc, long sC, long sCT, long ldcT,
                int M, int N, int K, int act)
{
  __shared__ __align__(16) __bf16 tA[2][TM][LDT];
  __shared__ __align__(16) __bf16 tB[2][TN][LDT];

  const int tid  = threadIdx.x;
  const int lane = tid & 31;
  const int wid  = tid >> 5;
  const int wm   = wid & 3;
  const int wn   = wid >> 2;
  const int lr   = lane & 15;
  const int lh   = lane >> 4;

  const int bm = blockIdx.y * TM;
  const int bn = blockIdx.x * TN;
  const int bz = blockIdx.z;

  const __bf16* Ab = A  + (long)bz * sA;
  const __bf16* Bb = Bw + (long)bz * sB;

#if SA_TDM
  unsigned int ldsA[2], ldsB[2];
  ldsA[0] = (unsigned int)(uintptr_t)(&tA[0][0][0]);
  ldsA[1] = (unsigned int)(uintptr_t)(&tA[1][0][0]);
  ldsB[0] = (unsigned int)(uintptr_t)(&tB[0][0][0]);
  ldsB[1] = (unsigned int)(uintptr_t)(&tB[1][0][0]);
  // prologue: fetch tile 0 into buffer 0
  if (wid == 0) {
    sa_tdm_load(ldsA[0], Ab + (long)bm * lda, TM, (unsigned int)lda);
    sa_tdm_load(ldsB[0], Bb + (long)bn * ldb, TN, (unsigned int)ldb);
    __builtin_amdgcn_s_wait_tensorcnt(0);
  }
  __syncthreads();
#endif

  v8f acc[2][2] = {};
  int cur = 0;

  for (int k0 = 0; k0 < K; k0 += TK) {
#if SA_TDM
    // kick off DMA for the next tile into the other buffer (no wait yet)
    if (wid == 0 && (k0 + TK) < K) {
      sa_tdm_load(ldsA[cur ^ 1], Ab + (long)bm * lda + (k0 + TK), TM, (unsigned int)lda);
      sa_tdm_load(ldsB[cur ^ 1], Bb + (long)bn * ldb + (k0 + TK), TN, (unsigned int)ldb);
    }
#else
    // fallback: vectorized 16B bf16 loads into buffer 0
    for (int idx = tid; idx < TM * 4; idx += 256) {
      int r = idx >> 2, c = (idx & 3) * 8;
      *(v8bf*)(&tA[0][r][c]) = *(const v8bf*)(Ab + (long)(bm + r) * lda + k0 + c);
    }
    for (int idx = tid; idx < TN * 4; idx += 256) {
      int r = idx >> 2, c = (idx & 3) * 8;
      *(v8bf*)(&tB[0][r][c]) = *(const v8bf*)(Bb + (long)(bn + r) * ldb + k0 + c);
    }
    __syncthreads();
#endif

    // Fragments per ISA VGPR layouts:
    // A 16x32 bf16: lane<16 -> M=lane,    K={0..7,16..23}
    //               lane>=16-> M=lane-16, K={8..15,24..31}
    // B 32x16 bf16: lane<16 -> N=lane, K={0..7,16..23} ; lane>=16 -> N=lane-16, K={8..15,24..31}
    const __bf16 (*bufA)[LDT] = tA[SA_TDM ? cur : 0];
    const __bf16 (*bufB)[LDT] = tB[SA_TDM ? cur : 0];
    v16bf afrag[2], bfrag[2];
    const int klo = lh * 8;
    #pragma unroll
    for (int s = 0; s < 2; ++s) {
      const __bf16* pa = &bufA[wm * 32 + s * 16 + lr][0];
      v8bf alo = *(const v8bf*)(pa + klo);
      v8bf ahi = *(const v8bf*)(pa + klo + 16);
      #pragma unroll
      for (int e = 0; e < 8; ++e) { afrag[s][e] = alo[e]; afrag[s][8 + e] = ahi[e]; }

      const __bf16* pb = &bufB[wn * 32 + s * 16 + lr][lh * 8];
      v8bf blo = *(const v8bf*)(pb);
      v8bf bhi = *(const v8bf*)(pb + 16);
      #pragma unroll
      for (int e = 0; e < 8; ++e) { bfrag[s][e] = blo[e]; bfrag[s][8 + e] = bhi[e]; }
    }

    #pragma unroll
    for (int i = 0; i < 2; ++i)
      #pragma unroll
      for (int j = 0; j < 2; ++j)
        acc[i][j] = __builtin_amdgcn_wmma_f32_16x16x32_bf16(
            false, afrag[i], false, bfrag[j], (short)0, acc[i][j], false, false);

#if SA_TDM
    // next buffer must be complete before anyone proceeds past the barrier
    if (wid == 0) __builtin_amdgcn_s_wait_tensorcnt(0);
    __syncthreads();
    cur ^= 1;
#else
    __syncthreads();
#endif
  }

  // Epilogue: C/D layout -> M = r + 8*lh, N = lr
  #pragma unroll
  for (int i = 0; i < 2; ++i) {
    #pragma unroll
    for (int j = 0; j < 2; ++j) {
      const int mb = bm + wm * 32 + i * 16 + lh * 8;
      const int nc = bn + wn * 32 + j * 16 + lr;
      const float bv = bias  ? bias[nc]  : 0.0f;
      const float rc = r1col ? r1col[nc] : 0.0f;
      #pragma unroll
      for (int r = 0; r < 8; ++r) {
        const int m = mb + r;
        float v = acc[i][j][r] + bv;
        if (r1row) v += r1row[(long)bz * 512 + m] * rc;
        if (act == 1)      v = tanhf(v);
        else if (act == 2) v = (v >= 0.0f) ? v : 0.01f * v;
        if (Cf) Cf[(long)bz * sC + (long)m * ldc + nc] = v;
        if (Cb) Cb[(long)bz * sC + (long)m * ldc + nc] = (__bf16)v;
        if (CbT) {
          int tb = m >> 9, tl = m & 511;
          CbT[(long)tb * sCT + (long)nc * ldcT + tl] = (__bf16)v;
        }
      }
    }
  }
}

// ---------------------------------------------------------------------------
// split x -> struct(bf16, row-major), sem(bf16, per-batch transposed [n][t]),
// and f_i_root = struct . w_root (fp32)
// ---------------------------------------------------------------------------
__global__ __launch_bounds__(256)
void sa_split(const float* __restrict__ x, const float* __restrict__ wroot,
              __bf16* __restrict__ structB, __bf16* __restrict__ semT,
              float* __restrict__ firoot)
{
  __shared__ float red[256];
  const int token = blockIdx.x;
  const int k = threadIdx.x;
  const float* xb = x + (long)token * 512;
  // sem[0:128]=x[0:128], sem[128:256]=x[256:384]
  // struct[0:128]=x[128:256], struct[128:256]=x[384:512]
  float sv  = xb[(k < 128) ? k : (k + 128)];
  float stv = xb[(k < 128) ? (k + 128) : (k + 256)];
  structB[(long)token * 256 + k] = (__bf16)stv;
  const int b = token >> 9, t = token & 511;
  semT[(long)b * (256 * 512) + (long)k * 512 + t] = (__bf16)sv;
  red[k] = stv * wroot[k];
  __syncthreads();
  for (int off = 128; off > 0; off >>= 1) {
    if (k < off) red[k] += red[k + off];
    __syncthreads();
  }
  if (k == 0) firoot[token] = red[0];
}

// fp32 -> bf16 (n % 4 == 0)
__global__ void sa_cvt(const float* __restrict__ src, __bf16* __restrict__ dst, long n)
{
  long i = ((long)blockIdx.x * 256 + threadIdx.x) * 4;
  if (i < n) {
    float4 v = *(const float4*)(src + i);
    dst[i + 0] = (__bf16)v.x; dst[i + 1] = (__bf16)v.y;
    dst[i + 2] = (__bf16)v.z; dst[i + 3] = (__bf16)v.w;
  }
}

// W_fij (256x256) transpose-convert: dst[k][d] = src[d][k]
__global__ void sa_cvtT256(const float* __restrict__ src, __bf16* __restrict__ dst)
{
  int kq = blockIdx.x, d = threadIdx.x;
  dst[(long)kq * 256 + d] = (__bf16)src[(long)d * 256 + kq];
}

// A = exp(f_ij) masked (in place on F, fp32)
__global__ void sa_buildA(float* __restrict__ F, const int* __restrict__ seq)
{
  long idx = (long)blockIdx.x * 256 + threadIdx.x;
  int b = (int)(idx >> 18);
  int rem = (int)(idx & (SA_TT - 1));
  int i = rem >> 9, j = rem & 511;
  int len = seq[b];
  float f = F[idx];
  F[idx] = (i != j && i < len && j < len) ? expf(f) : 0.0f;
}

__global__ void sa_aroot(const float* __restrict__ firoot, const int* __restrict__ seq,
                         float* __restrict__ ar)
{
  int idx = blockIdx.x * 256 + threadIdx.x;
  int b = idx >> 9, t = idx & 511;
  ar[idx] = (t < seq[b]) ? expf(firoot[idx]) : 0.0f;
}

// col_sum[b][j] = sum_i A[b][i][j]
__global__ void sa_colsum(const float* __restrict__ A, float* __restrict__ cs)
{
  int blk = blockIdx.x;
  int b = blk >> 1;
  int j = ((blk & 1) << 8) + threadIdx.x;
  const float* Ab = A + (long)b * SA_TT;
  float s = 0.0f;
  for (int i = 0; i < SA_T; ++i) s += Ab[(long)i * SA_T + j];
  cs[b * SA_T + j] = s;
}

__global__ void sa_buildL(const float* __restrict__ A, const float* __restrict__ cs,
                          const float* __restrict__ ar, const int* __restrict__ seq,
                          float* __restrict__ L)
{
  long idx = (long)blockIdx.x * 256 + threadIdx.x;
  int b = (int)(idx >> 18);
  int rem = (int)(idx & (SA_TT - 1));
  int i = rem >> 9, j = rem & 511;
  int len = seq[b];
  float mj = (j < len) ? 0.0f : 1.0f;
  float v;
  if (i == 0) v = ar[b * SA_T + j] + ((j == 0) ? mj : 0.0f);
  else        v = ((i == j) ? (cs[b * SA_T + j] + mj) : 0.0f) - A[idx];
  L[idx] = v;
}

// In-place Gauss-Jordan inverse, one block (512 threads) per batch matrix.
__global__ __launch_bounds__(512)
void sa_invert(float* __restrict__ Mx)
{
  __shared__ float rowk[512];
  __shared__ float fac[512];
  const int tid = threadIdx.x;
  float* Mb = Mx + (long)blockIdx.x * SA_TT;
  for (int k = 0; k < 512; ++k) {
    float piv  = Mb[(long)k * 512 + k];
    float invp = 1.0f / piv;
    float v = Mb[(long)k * 512 + tid];
    v = (tid == k) ? invp : v * invp;
    rowk[tid] = v;
    fac[tid]  = (tid == k) ? 0.0f : Mb[(long)tid * 512 + k];
    __syncthreads();
    Mb[(long)k * 512 + tid] = rowk[tid];
    for (int i = 0; i < 512; ++i) {
      if (i == k) continue;
      float f = fac[i];
      long off = (long)i * 512 + tid;
      float cur = Mb[off];
      Mb[off] = (tid == k) ? (-f * invp) : (cur - f * rowk[tid]);
    }
    __syncthreads();
  }
}

// p_ij = A * (nz_j*Linv[j][j] - nz_i*Linv[j][i]); bf16 p_ij + bf16 p_ij^T,
// fp32 p_ij_root rows 1..T
__global__ void sa_pij(const float* __restrict__ A, const float* __restrict__ Linv,
                       __bf16* __restrict__ pijB, __bf16* __restrict__ pijT,
                       float* __restrict__ prout)
{
  long idx = (long)blockIdx.x * 256 + threadIdx.x;
  int b = (int)(idx >> 18);
  int rem = (int)(idx & (SA_TT - 1));
  int i = rem >> 9, j = rem & 511;
  const float* Lb = Linv + (long)b * SA_TT;
  float a   = A[idx];
  float dj  = (j > 0) ? Lb[(long)j * 512 + j] : 0.0f;
  float lji = (i > 0) ? Lb[(long)j * 512 + i] : 0.0f;
  float p = a * (dj - lji);
  pijB[idx] = (__bf16)p;
  pijT[(long)b * SA_TT + (long)j * 512 + i] = (__bf16)p;
  prout[(long)b * (513 * 512) + (long)(i + 1) * 512 + j] = p;
}

// p_root = A_root * Linv[:, :, 0]; fp32 p_ij_root row 0
__global__ void sa_proot(const float* __restrict__ ar, const float* __restrict__ Linv,
                         float* __restrict__ proot, float* __restrict__ prout)
{
  int idx = blockIdx.x * 256 + threadIdx.x;
  int b = idx >> 9, t = idx & 511;
  float pr = ar[idx] * Linv[(long)b * SA_TT + (long)t * 512 + 0];
  proot[idx] = pr;
  prout[(long)b * (513 * 512) + t] = pr;
}

// ---------------------------------------------------------------------------
extern "C" void kernel_launch(void* const* d_in, const int* in_sizes, int n_in,
                              void* d_out, int out_size, void* d_ws, size_t ws_size,
                              hipStream_t stream)
{
  const float* x       = (const float*)d_in[0];
  const int*   seq     = (const int*)  d_in[1];
  const float* W_tp    = (const float*)d_in[2];
  const float* b_tp    = (const float*)d_in[3];
  const float* W_tc    = (const float*)d_in[4];
  const float* b_tc    = (const float*)d_in[5];
  const float* W_fij   = (const float*)d_in[6];
  const float* w_root  = (const float*)d_in[7];
  const float* rootemb = (const float*)d_in[8];
  const float* W_r     = (const float*)d_in[9];
  const float* b_r     = (const float*)d_in[10];
  const float* W_perc  = (const float*)d_in[11];
  const float* b_perc  = (const float*)d_in[12];

  float* out = (float*)d_out;
  float* cur = (float*)d_ws;

  const long SZ_TOK = (long)SA_BT * SA_S;   // 4,194,304
  const long SZ_MAT = (long)SA_B * SA_TT;   // 8,388,608

  // fp32 workspace
  float* Amat   = cur; cur += SZ_MAT;       // f_ij -> A (in place)
  float* Lmat   = cur; cur += SZ_MAT;       // L_bar -> Linv (in place)
  float* csum   = cur; cur += SA_BT;
  float* aroot  = cur; cur += SA_BT;
  float* proot  = cur; cur += SA_BT;
  float* firoot = cur; cur += SA_BT;
  // bf16 workspace (2 elems per float slot; all counts even)
  __bf16* structB = (__bf16*)cur; cur += SZ_TOK / 2;    // [BT,256]
  __bf16* semT    = (__bf16*)cur; cur += SZ_TOK / 2;    // [B,256,512]
  __bf16* tpB     = (__bf16*)cur; cur += SZ_TOK / 2;    // [BT,256]
  __bf16* tcB     = (__bf16*)cur; cur += SZ_TOK / 2;    // [BT,256]
  __bf16* tpWB    = (__bf16*)cur; cur += SZ_TOK / 2;    // [BT,256]
  __bf16* pijB    = (__bf16*)cur; cur += SZ_MAT / 2;    // [B,T,T]
  __bf16* pijT    = (__bf16*)cur; cur += SZ_MAT / 2;    // [B,T,T] transposed
  __bf16* PCB     = (__bf16*)cur; cur += SZ_MAT / 2;    // [BT,512] parents|children
  __bf16* OCB     = (__bf16*)cur; cur += SZ_MAT / 2;    // [BT,512] out0|cp
  __bf16* OT      = (__bf16*)cur; cur += SZ_TOK / 2;    // [B,256,512] out0^T
  __bf16* WtpB    = (__bf16*)cur; cur += 65536 / 2;
  __bf16* WtcB    = (__bf16*)cur; cur += 65536 / 2;
  __bf16* WfijT   = (__bf16*)cur; cur += 65536 / 2;     // transposed (k,d)
  __bf16* WrB     = (__bf16*)cur; cur += 131072 / 2;
  __bf16* WpercB  = (__bf16*)cur; cur += 131072 / 2;

  float* out_main = out;                 // (B,T,256)
  float* prout    = out + SZ_TOK;        // p_ij_root (B,513,512)

  const long ST_A  = (long)SA_T * SA_S;  // 131072
  const long ST_TT = (long)SA_TT;        // 262144

  // 1) split + f_i_root; weight conversions
  sa_split<<<SA_BT, 256, 0, stream>>>(x, w_root, structB, semT, firoot);
  sa_cvt<<<64, 256, 0, stream>>>(W_tp, WtpB, 65536);
  sa_cvt<<<64, 256, 0, stream>>>(W_tc, WtcB, 65536);
  sa_cvtT256<<<256, 256, 0, stream>>>(W_fij, WfijT);
  sa_cvt<<<128, 256, 0, stream>>>(W_r, WrB, 131072);
  sa_cvt<<<128, 256, 0, stream>>>(W_perc, WpercB, 131072);

  // 2) tp = tanh(struct @ W_tp^T + b_tp)   -> bf16
  sa_gemm_nt<<<dim3(SA_S / TN, SA_BT / TM, 1), 256, 0, stream>>>(
      structB, SA_S, 0, WtpB, SA_S, 0, b_tp, nullptr, nullptr,
      nullptr, tpB, nullptr, SA_S, 0, 0, 0, SA_BT, SA_S, SA_S, 1);
  // 3) tc = tanh(struct @ W_tc^T + b_tc)   -> bf16
  sa_gemm_nt<<<dim3(SA_S / TN, SA_BT / TM, 1), 256, 0, stream>>>(
      structB, SA_S, 0, WtcB, SA_S, 0, b_tc, nullptr, nullptr,
      nullptr, tcB, nullptr, SA_S, 0, 0, 0, SA_BT, SA_S, SA_S, 1);
  // 4) tpW = tp @ W_fij                    -> bf16  (B = WfijT, NT)
  sa_gemm_nt<<<dim3(SA_S / TN, SA_BT / TM, 1), 256, 0, stream>>>(
      tpB, SA_S, 0, WfijT, SA_S, 0, nullptr, nullptr, nullptr,
      nullptr, tpWB, nullptr, SA_S, 0, 0, 0, SA_BT, SA_S, SA_S, 0);
  // 5) f_ij[b] = tpW[b] @ tc[b]^T          -> fp32 Amat (batched)
  sa_gemm_nt<<<dim3(SA_T / TN, SA_T / TM, SA_B), 256, 0, stream>>>(
      tpWB, SA_S, ST_A, tcB, SA_S, ST_A, nullptr, nullptr, nullptr,
      Amat, nullptr, nullptr, SA_T, ST_TT, 0, 0, SA_T, SA_T, SA_S, 0);

  // 6) matrix-tree prep
  sa_aroot <<<SA_BT / 256, 256, 0, stream>>>(firoot, seq, aroot);
  sa_buildA<<<(int)(SZ_MAT / 256), 256, 0, stream>>>(Amat, seq);
  sa_colsum<<<SA_B * 2, 256, 0, stream>>>(Amat, csum);
  sa_buildL<<<(int)(SZ_MAT / 256), 256, 0, stream>>>(Amat, csum, aroot, seq, Lmat);

  // 7) Linv = inv(L_bar), in place
  sa_invert<<<SA_B, 512, 0, stream>>>(Lmat);

  // 8) marginals (bf16 p_ij / p_ij^T + fp32 p_ij_root to d_out)
  sa_pij  <<<(int)(SZ_MAT / 256), 256, 0, stream>>>(Amat, Lmat, pijB, pijT, prout);
  sa_proot<<<SA_BT / 256, 256, 0, stream>>>(aroot, Lmat, proot, prout);

  // 9) children[b] = p_ij[b] @ sem[b]      -> PCB[:,256:512]
  sa_gemm_nt<<<dim3(SA_S / TN, SA_T / TM, SA_B), 256, 0, stream>>>(
      pijB, SA_T, ST_TT, semT, SA_T, ST_A, nullptr, nullptr, nullptr,
      nullptr, PCB + 256, nullptr, 512, ST_TT, 0, 0, SA_T, SA_S, SA_T, 0);
  // 10) parents[b] = p_ij[b]^T @ sem[b] + p_root (x) root_emb -> PCB[:,0:256]
  sa_gemm_nt<<<dim3(SA_S / TN, SA_T / TM, SA_B), 256, 0, stream>>>(
      pijT, SA_T, ST_TT, semT, SA_T, ST_A, nullptr, proot, rootemb,
      nullptr, PCB, nullptr, 512, ST_TT, 0, 0, SA_T, SA_S, SA_T, 0);

  // 11) out0 = leaky(PC @ W_r^T + b_r)     -> OCB[:,0:256] and OT (transposed)
  sa_gemm_nt<<<dim3(SA_S / TN, SA_BT / TM, 1), 256, 0, stream>>>(
      PCB, 512, 0, WrB, 512, 0, b_r, nullptr, nullptr,
      nullptr, OCB, OT, 512, 0, ST_A, 512, SA_BT, SA_S, 512, 2);
  // 12) cp[b] = p_ij[b] @ out0[b]          -> OCB[:,256:512]
  sa_gemm_nt<<<dim3(SA_S / TN, SA_T / TM, SA_B), 256, 0, stream>>>(
      pijB, SA_T, ST_TT, OT, SA_T, ST_A, nullptr, nullptr, nullptr,
      nullptr, OCB + 256, nullptr, 512, ST_TT, 0, 0, SA_T, SA_S, SA_T, 0);
  // 13) out = leaky(OC @ W_perc^T + b_perc) -> d_out (fp32)
  sa_gemm_nt<<<dim3(SA_S / TN, SA_BT / TM, 1), 256, 0, stream>>>(
      OCB, 512, 0, WpercB, 512, 0, b_perc, nullptr, nullptr,
      out_main, nullptr, nullptr, SA_S, 0, 0, 0, SA_BT, SA_S, 512, 2);
}
